// CustomLSTM_11175504904490
// MI455X (gfx1250) — matrix operand order repaired and checked by reading, not compiled, not run on hardware
//
#include <hip/hip_runtime.h>
#include <hip/hip_fp16.h>

// ---------------------------------------------------------------------------
// Bidirectional LSTM, persistent-kernel design for MI455X (gfx1250, wave32).
//
//   - 64 workgroups (32 per direction), each owns 16 hidden units.
//   - W_ih / W_hh column slices live in LDS as f16 for the entire sequence.
//   - Per-step gates GEMM uses v_wmma_f32_16x16x32_f16 (f32 accumulate),
//     software-pipelined (double-buffered fragments) to hide LDS/L2 latency
//     inside the sequential recurrence.
//   - h is double-buffered in workspace (f16); c stays in registers.
//   - Device-scope step barrier: monotonic atomic counter per direction.
// ---------------------------------------------------------------------------

#define SEQL   512
#define BATCH  64
#define INP    512
#define HIDDEN 512

typedef __attribute__((ext_vector_type(16))) _Float16     v16h;
typedef __attribute__((ext_vector_type(8)))  float        v8f;
typedef __attribute__((ext_vector_type(4)))  unsigned int v4u;

union FragH {        // 16 halves = one WMMA A or B operand (8 VGPRs)
  v16h h;
  v4u  q[2];
};

#define WPITCH 520   // halves per weight row in LDS: 1040B rows (16B aligned,
                     // 260-dword stride -> 16 lanes land on distinct banks)
#define GPITCH 65    // floats per gates row in LDS (odd pitch, conflict-free)

#define SMEM_BYTES (2 * 64 * WPITCH * 2 /*Wih+Whh f16*/ \
                    + 64 * GPITCH * 4   /*gates f32*/   \
                    + 64 * 4            /*bias f32*/)

#define KGROUPS 32   // 16 groups of K=32 for x-part + 16 for h-part

__device__ __forceinline__ float sigmoidf_(float x) {
  return 1.0f / (1.0f + __expf(-x));
}

// ---------------------------------------------------------------------------
// Init: convert x to f16 once, zero the h double-buffers, reset barrier
// counters.  Runs every call so graph replays are deterministic.
// ---------------------------------------------------------------------------
__global__ void lstm_bidir_init_kernel(const float* __restrict__ x,
                                       _Float16* __restrict__ xh,
                                       _Float16* __restrict__ hbuf,
                                       unsigned* __restrict__ cnt,
                                       int nx, int nh) {
  const int stride = gridDim.x * blockDim.x;
  const int tid    = blockIdx.x * blockDim.x + threadIdx.x;
  for (int i = tid; i < nx; i += stride) xh[i] = (_Float16)x[i];
  for (int i = tid; i < nh; i += stride) hbuf[i] = (_Float16)0.0f;
  if (tid < 16) cnt[tid] = 0u;
}

// ---------------------------------------------------------------------------
// Persistent bidirectional LSTM kernel.
// grid = 64 blocks x 256 threads.  blockIdx>>5 = direction, &31 = hidden slice.
// ---------------------------------------------------------------------------
__global__ void __launch_bounds__(256, 1)
lstm_bidir_persistent_kernel(const _Float16* __restrict__ xh,
                             _Float16* __restrict__ hbuf,  // [2 dir][2 ph][64][512]
                             const float* __restrict__ Wih_f,
                             const float* __restrict__ Whh_f,
                             const float* __restrict__ bih_f,
                             const float* __restrict__ bhh_f,
                             const float* __restrict__ Wih_b,
                             const float* __restrict__ Whh_b,
                             const float* __restrict__ bih_b,
                             const float* __restrict__ bhh_b,
                             float* __restrict__ out,
                             unsigned* __restrict__ cnt) {
  extern __shared__ char smem[];
  _Float16* WihL   = (_Float16*)smem;               // [64][WPITCH] f16
  _Float16* WhhL   = WihL + 64 * WPITCH;            // [64][WPITCH] f16
  float*    gatesL = (float*)(WhhL + 64 * WPITCH);  // [64][GPITCH] f32
  float*    biasL  = gatesL + 64 * GPITCH;          // [64] f32

  const int tid   = threadIdx.x;
  const int dir   = blockIdx.x >> 5;   // 0 = forward, 1 = backward
  const int slice = blockIdx.x & 31;   // which 16 hidden units
  const int hbase = slice * 16;

  const float* Wih = dir ? Wih_b : Wih_f;
  const float* Whh = dir ? Whh_b : Whh_f;
  const float* bih = dir ? bih_b : bih_f;
  const float* bhh = dir ? bhh_b : bhh_f;

  // ---- one-time: stage this block's weight slice into LDS as f16 ----------
  // local gate column c in 0..63 maps to global gate row g*512 + hbase + j,
  // g = c>>4 (i,f,g,o), j = c&15.
  for (int e = tid; e < 64 * 512; e += 256) {
    const int c = e >> 9;
    const int k = e & 511;
    const int grow = ((c >> 4) << 9) + hbase + (c & 15);
    WihL[c * WPITCH + k] = (_Float16)Wih[grow * 512 + k];
    WhhL[c * WPITCH + k] = (_Float16)Whh[grow * 512 + k];
  }
  if (tid < 64) {
    const int grow = ((tid >> 4) << 9) + hbase + (tid & 15);
    biasL[tid] = bih[grow] + bhh[grow];
  }
  __syncthreads();

  // ---- wave -> tile mapping ------------------------------------------------
  const int lane = tid & 31;
  const int wid  = tid >> 5;        // 0..7
  const int mi   = wid >> 1;        // batch tile 0..3  (rows mi*16..+15)
  const int nb   = (wid & 1) * 2;   // this wave owns gate-col tiles nb, nb+1

  const int arow = mi * 16 + (lane & 15);  // A-matrix row for this lane
  const int sel  = lane >> 4;              // half-wave selector (K grouping)

  // Per-lane base pointers into the LDS weight slabs (col0 / col1 rows).
  const _Float16* W0ih = WihL + ((nb + 0) * 16 + (lane & 15)) * WPITCH + sel * 8;
  const _Float16* W1ih = WihL + ((nb + 1) * 16 + (lane & 15)) * WPITCH + sel * 8;
  const _Float16* W0hh = WhhL + ((nb + 0) * 16 + (lane & 15)) * WPITCH + sel * 8;
  const _Float16* W1hh = WhhL + ((nb + 1) * 16 + (lane & 15)) * WPITCH + sel * 8;

  // ---- per-thread persistent cell state -----------------------------------
  // thread owns batch row m = tid&63 and hidden units j0, j0+4, j0+8, j0+12
  const int m  = tid & 63;
  const int j0 = tid >> 6;
  float creg[4] = {0.f, 0.f, 0.f, 0.f};

  _Float16* hdir  = hbuf + (size_t)dir * 2 * BATCH * HIDDEN;
  unsigned* mycnt = cnt + dir;

  for (int t = 0; t < SEQL; ++t) {
    const int tin = dir ? (SEQL - 1 - t) : t;      // scanned input index
    // per-lane A-row base pointers for this step
    const _Float16* Axr = xh + (size_t)tin * BATCH * INP + arow * INP + sel * 8;
    const _Float16* Ahr = hdir + (size_t)(t & 1) * BATCH * HIDDEN
                          + arow * HIDDEN + sel * 8;
    _Float16* Hw = hdir + (size_t)((t + 1) & 1) * BATCH * HIDDEN;

    // prefetch next step's activations (64KB slab; 256 threads cover it)
    if (t + 1 < SEQL) {
      const int tnx = dir ? (SEQL - 2 - t) : (t + 1);
      __builtin_prefetch(xh + (size_t)tnx * BATCH * INP + tid * 128, 0, 0);
    }

    v8f acc0 = {};
    v8f acc1 = {};

    // ---- gates = x_t*W_ih^T + h_{t-1}*W_hh^T  (K = 1024 total) -------------
    // Single fully-unrolled 32-group loop; groups 0..15 read x_t / W_ih,
    // groups 16..31 read h / W_hh (selection folds away under full unroll).
    // 2-stage software pipeline: fragments for group i+1 are loaded before
    // the WMMAs of group i, so LDS/L2 latency overlaps matrix issue.
    FragH af[2], b0f[2], b1f[2];

    auto load_group = [&](int i, FragH& a, FragH& b0, FragH& b1) {
      const int kb = (i & 15) * 32;
      const _Float16* Ab = (i < 16) ? Axr : Ahr;
      const _Float16* p0 = ((i < 16) ? W0ih : W0hh) + kb;
      const _Float16* p1 = ((i < 16) ? W1ih : W1hh) + kb;
      a.q[0]  = *(const v4u*)(Ab + kb);        // K = kb+{0..7 | 8..15}
      a.q[1]  = *(const v4u*)(Ab + kb + 16);   // K = kb+16+{0..7 | 8..15}
      b0.q[0] = ((const v4u*)p0)[0];           // K = kb+sel*8+{0..7}, kb+16+sel*8+{0..7}
      b0.q[1] = ((const v4u*)p0)[2];
      b1.q[0] = ((const v4u*)p1)[0];
      b1.q[1] = ((const v4u*)p1)[2];
    };

    load_group(0, af[0], b0f[0], b1f[0]);
#pragma unroll
    for (int i = 0; i < KGROUPS; ++i) {
      const int cur = i & 1;
      const int nxt = cur ^ 1;
      if (i + 1 < KGROUPS) load_group(i + 1, af[nxt], b0f[nxt], b1f[nxt]);
      acc0 = __builtin_amdgcn_wmma_f32_16x16x32_f16(
          false, af[cur].h, false, b0f[cur].h, (short)0, acc0, false, false);
      acc1 = __builtin_amdgcn_wmma_f32_16x16x32_f16(
          false, af[cur].h, false, b1f[cur].h, (short)0, acc1, false, false);
    }

    // ---- spill gate tiles to LDS (C layout: vgpr v -> row, lane%16 -> col) -
    {
      const int crow = mi * 16 + sel * 8;
      const int cc0  = (nb + 0) * 16 + (lane & 15);
      const int cc1  = (nb + 1) * 16 + (lane & 15);
#pragma unroll
      for (int v = 0; v < 8; ++v) {
        gatesL[(crow + v) * GPITCH + cc0] = acc0[v];
        gatesL[(crow + v) * GPITCH + cc1] = acc1[v];
      }
    }
    __syncthreads();

    // ---- fused LSTM cell update (c in registers) ---------------------------
#pragma unroll
    for (int u = 0; u < 4; ++u) {
      const int j = j0 + 4 * u;                      // hidden unit 0..15
      const float gi = gatesL[m * GPITCH + j]      + biasL[j];
      const float gf = gatesL[m * GPITCH + 16 + j] + biasL[16 + j];
      const float gg = gatesL[m * GPITCH + 32 + j] + biasL[32 + j];
      const float go = gatesL[m * GPITCH + 48 + j] + biasL[48 + j];
      const float ii = sigmoidf_(gi);
      const float ff = sigmoidf_(gf);
      const float g  = tanhf(gg);
      const float oo = sigmoidf_(go);
      const float cc = ff * creg[u] + ii * g;
      creg[u] = cc;
      const float hh = oo * tanhf(cc);
      Hw[m * HIDDEN + hbase + j] = (_Float16)hh;
      // backward outputs intentionally NOT flipped (replicates reference)
      out[((size_t)t * BATCH + m) * (2 * HIDDEN) + dir * HIDDEN + hbase + j] = hh;
    }
    __syncthreads();     // gatesL safe to rewrite after this point
    __threadfence();     // publish Hw before signaling

    // ---- device-scope step barrier over this direction's 32 workgroups ----
    // Monotonic counter: a wave can only start round r+1 after cnt >= (r+1)*32,
    // so rounds are totally ordered and no reset is needed intra-kernel.
    if (tid == 0) {
      const unsigned old =
          __hip_atomic_fetch_add(mycnt, 1u, __ATOMIC_ACQ_REL,
                                 __HIP_MEMORY_SCOPE_AGENT);
      const unsigned target = (old / 32u + 1u) * 32u;
      while (__hip_atomic_load(mycnt, __ATOMIC_ACQUIRE,
                               __HIP_MEMORY_SCOPE_AGENT) < target) {
        __builtin_amdgcn_s_sleep(8);
      }
    }
    __syncthreads();
    __threadfence();     // acquire side: see other blocks' Hw next iteration
  }
}

// ---------------------------------------------------------------------------
extern "C" void kernel_launch(void* const* d_in, const int* in_sizes, int n_in,
                              void* d_out, int out_size, void* d_ws,
                              size_t ws_size, hipStream_t stream) {
  (void)in_sizes; (void)n_in; (void)out_size; (void)ws_size;

  const float* x     = (const float*)d_in[0];
  const float* Wih_f = (const float*)d_in[1];
  const float* Whh_f = (const float*)d_in[2];
  const float* bih_f = (const float*)d_in[3];
  const float* bhh_f = (const float*)d_in[4];
  const float* Wih_b = (const float*)d_in[5];
  const float* Whh_b = (const float*)d_in[6];
  const float* bih_b = (const float*)d_in[7];
  const float* bhh_b = (const float*)d_in[8];
  float* out = (float*)d_out;

  // workspace layout
  const int NX = SEQL * BATCH * INP;              // 16,777,216 halves
  const int NH = 2 * 2 * BATCH * HIDDEN;          // 131,072 halves
  _Float16* xh   = (_Float16*)d_ws;
  _Float16* hbuf = xh + NX;
  unsigned* cnt  = (unsigned*)(hbuf + NH);

  hipFuncSetAttribute((const void*)lstm_bidir_persistent_kernel,
                      hipFuncAttributeMaxDynamicSharedMemorySize, SMEM_BYTES);

  lstm_bidir_init_kernel<<<512, 256, 0, stream>>>(x, xh, hbuf, cnt, NX, NH);
  lstm_bidir_persistent_kernel<<<64, 256, SMEM_BYTES, stream>>>(
      xh, hbuf, Wih_f, Whh_f, bih_f, bhh_f, Wih_b, Whh_b, bih_b, bhh_b, out,
      cnt);
}
